// DownNet_19524921328200
// MI455X (gfx1250) — matrix operand order repaired and checked by reading, not compiled, not run on hardware
//
#include <hip/hip_runtime.h>
#include <hip/hip_bf16.h>
#include <cstddef>

// ---------------------------------------------------------------------------
// CDNA5 wave32 WMMA types: 16x16x32 bf16 -> f32
// ---------------------------------------------------------------------------
typedef __attribute__((ext_vector_type(16))) __bf16 v16bf;  // 8 VGPRs (bf16 pairs)
typedef __attribute__((ext_vector_type(8)))  __bf16 v8bf;   // 16B load
typedef __attribute__((ext_vector_type(8)))  float  v8f;    // 16x16 f32 C/D tile
typedef __attribute__((ext_vector_type(4)))  float  f32x4;
typedef __attribute__((ext_vector_type(4)))  int    v4i;

// Pointer types matching the async-copy builtin signature:
//   param0: v4i in AS(1) (global, printed '__device__'), param1: v4i in AS(3) (LDS)
typedef __attribute__((address_space(1))) v4i as1_v4i;
typedef __attribute__((address_space(3))) v4i as3_v4i;

// gfx1250 async global->LDS copy path (ASYNCcnt-tracked), with fallback
#if defined(__gfx1250__) && \
    __has_builtin(__builtin_amdgcn_global_load_async_to_lds_b128) && \
    __has_builtin(__builtin_amdgcn_s_wait_asynccnt)
#define USE_ASYNC_LDS 1
#else
#define USE_ASYNC_LDS 0
#endif

static const int kLEVELS[5] = {40962, 10242, 2562, 642, 162};
static const int kCHS[6]    = {64, 32, 64, 128, 256, 512};

// ---------------------------------------------------------------------------
// f32 -> bf16 conversion (x once, each weight matrix once per launch)
// ---------------------------------------------------------------------------
__global__ __launch_bounds__(256) void cvt_bf16(
    const float* __restrict__ src, __bf16* __restrict__ dst, int n)
{
    int i = blockIdx.x * 256 + threadIdx.x;
    if (i < n) dst[i] = (__bf16)src[i];
}

// ---------------------------------------------------------------------------
// onering conv as WMMA GEMM:  Y[N,Cout] = gather7(h,no)[N,7*Cin] * W^T + bias
// h and W bf16, f32 accumulation. Block = 8 waves: one 16-node tile,
// 8 x 16 output channels (grid.y spans Cout in 128-wide groups).
//
// Since K chunks are 32-aligned and Cin % 32 == 0, every 32-wide K chunk sits
// inside a single neighbor j => loop j outer, channels inner: one neighbor
// index load + one row base per j, zero per-chunk index bookkeeping.
//
// A-fragment (16-bit A 16x32, contiguous K): row = lane&15, K(e)=(lane>>4)*16+e
//   -> two contiguous 8-element bf16 runs per lane; wave 0 copies them straight
//      into the fragment-order LDS buffer via GLOBAL_LOAD_ASYNC_TO_LDS_B128.
// B-fragment (same K order as A): col = lane&15, K(e)=(lane>>4)*16+e
//   -> one 32B contiguous load from the pre-converted bf16 weight row.
// D transposed through LDS -> 2 x global_store_b128 per lane.
// ---------------------------------------------------------------------------
__global__ __launch_bounds__(256) void conv_onering_wmma(
    const __bf16* __restrict__ h_in, const int* __restrict__ no,
    const __bf16* __restrict__ W, const float* __restrict__ bias,
    float* __restrict__ y, int N, int Cin, int Cout)
{
    const int K7       = 7 * Cin;
    const int lane     = threadIdx.x & 31;
    const int wv       = threadIdx.x >> 5;
    const int tileBase = blockIdx.x * 16;
    const int colTile  = blockIdx.y * 8 + wv;
    const bool active  = (colTile * 16) < Cout;
    const int col      = colTile * 16 + (lane & 15);

    __shared__ __align__(32) __bf16 aFrag[512];   // 32 lanes x 16 bf16
    __shared__ float dTile[8 * 256];              // per-wave 16x16 f32 tile

    v8f acc = {0.f, 0.f, 0.f, 0.f, 0.f, 0.f, 0.f, 0.f};

    const int khalf = (lane >> 4) << 4;           // 0 or 16 (A layout)
    const int row   = lane & 15;

    int node = tileBase + row;
    if (node >= N) node = N - 1;                  // tail rows clamped; stores guarded
    const int* nb = no + 7 * node;

    // B pointer for this lane's 16 contiguous K values; advances 32/chunk
    const __bf16* wp = W + (size_t)col * K7 + ((lane >> 4) << 4);

    auto chunk = [&](const __bf16* srow, int cc) {
        if (wv == 0) {
            const __bf16* s0 = srow + cc + khalf;        // run0: e 0..7
            const __bf16* s1 = s0 + 8;                   // run1: e 8..15
            __bf16* d0 = aFrag + lane * 16;
#if USE_ASYNC_LDS
            __builtin_amdgcn_global_load_async_to_lds_b128(
                (as1_v4i*)s0, (as3_v4i*)d0, 0, 0);
            __builtin_amdgcn_global_load_async_to_lds_b128(
                (as1_v4i*)s1, (as3_v4i*)(d0 + 8), 0, 0);
            __builtin_amdgcn_s_wait_asynccnt(0);
#else
            v8bf lo = *(const v8bf*)s0;
            v8bf hi = *(const v8bf*)s1;
            *(v16bf*)d0 = __builtin_shufflevector(
                lo, hi, 0, 1, 2, 3, 4, 5, 6, 7, 8, 9, 10, 11, 12, 13, 14, 15);
#endif
        }
        __syncthreads();
        v16bf a = *(const v16bf*)(aFrag + lane * 16);
        if (active) {
            __builtin_prefetch(wp + 64, 0, 1);
            v16bf bv = *(const v16bf*)wp;
            acc = __builtin_amdgcn_wmma_f32_16x16x32_bf16(
                false, a, false, bv, (short)0, acc, false, false);
        }
        wp += 32;
        __syncthreads();
    };

    for (int j = 0; j < 7; ++j) {
        const __bf16* srow = h_in + (size_t)nb[j] * (size_t)Cin;
        int cc = 0;
        for (; cc + 64 <= Cin; cc += 64) { chunk(srow, cc); chunk(srow, cc + 32); }
        if (cc < Cin) chunk(srow, cc);   // Cin==32 remainder
    }

    // ---- epilogue: bias add, LDS transpose, coalesced b128 stores ----
    if (active) {
        float bval = bias[col];
        float* t = dTile + wv * 256;
        const int rtop = (lane >> 4) << 3;        // D layout: lanes>=16 hold M+8
        #pragma unroll
        for (int r = 0; r < 8; ++r)
            t[(r + rtop) * 16 + (lane & 15)] = acc[r] + bval;
    }
    __syncthreads();
    if (active) {
        const int orow = lane >> 1;
        const int half = (lane & 1) << 3;
        const int onode = tileBase + orow;
        if (onode < N) {
            const float* t = dTile + wv * 256 + orow * 16 + half;
            f32x4 o0 = *(const f32x4*)t;
            f32x4 o1 = *(const f32x4*)(t + 4);
            float* dst = y + (size_t)onode * Cout + colTile * 16 + half;
            *(f32x4*)dst       = o0;
            *(f32x4*)(dst + 4) = o1;
        }
    }
}

// ---------------------------------------------------------------------------
// Per-channel batch stats: stats[2c]=mean, stats[2c+1]=rsqrt(var+eps) (biased)
// ---------------------------------------------------------------------------
__global__ __launch_bounds__(256) void bn_stats(
    const float* __restrict__ y, int N, int C, float* __restrict__ stats)
{
    const int c = blockIdx.x;
    __shared__ float ssum[256];
    __shared__ float ssq[256];
    float s = 0.f, q = 0.f;
    for (int i = threadIdx.x; i < N; i += 256) {
        float v = y[(size_t)i * C + c];
        s += v; q += v * v;
    }
    ssum[threadIdx.x] = s; ssq[threadIdx.x] = q;
    __syncthreads();
    for (int off = 128; off > 0; off >>= 1) {
        if (threadIdx.x < off) {
            ssum[threadIdx.x] += ssum[threadIdx.x + off];
            ssq[threadIdx.x]  += ssq[threadIdx.x + off];
        }
        __syncthreads();
    }
    if (threadIdx.x == 0) {
        float inv_n = 1.0f / (float)N;
        float mean = ssum[0] * inv_n;
        float var  = ssq[0] * inv_n - mean * mean;
        stats[2 * c]     = mean;
        stats[2 * c + 1] = rsqrtf(var + 1e-5f);
    }
}

// BN apply + LeakyReLU, emitting the bf16 activation buffer for the next conv
__global__ __launch_bounds__(256) void bn_lrelu_bf16(
    const float* __restrict__ y, __bf16* __restrict__ hout, int N, int C,
    const float* __restrict__ stats, const float* __restrict__ gamma,
    const float* __restrict__ beta)
{
    int idx = blockIdx.x * 256 + threadIdx.x;
    if (idx >= N * C) return;
    int c = idx % C;
    float v = (y[idx] - stats[2 * c]) * stats[2 * c + 1] * gamma[c] + beta[c];
    hout[idx] = (__bf16)(v > 0.f ? v : 0.2f * v);
}

// ---------------------------------------------------------------------------
// 7-neighbor mean pool (bf16 in/out, f32 accumulate)
// ---------------------------------------------------------------------------
__global__ __launch_bounds__(256) void pool_mean7_bf16(
    const __bf16* __restrict__ h_in, const int* __restrict__ no,
    __bf16* __restrict__ h_out, int num, int C)
{
    int idx = blockIdx.x * 256 + threadIdx.x;
    if (idx >= num * C) return;
    int n = idx / C, c = idx - n * C;
    const int* nb = no + 7 * n;
    float s = 0.f;
    #pragma unroll
    for (int j = 0; j < 7; ++j) s += (float)h_in[(size_t)nb[j] * C + c];
    h_out[idx] = (__bf16)(s * (1.0f / 7.0f));
}

// ---------------------------------------------------------------------------
// outc: out[o] = bout[o] + dot(Wout[o,:], h.flatten())   (len = 162*512)
// ---------------------------------------------------------------------------
__global__ __launch_bounds__(256) void outc_dot(
    const __bf16* __restrict__ h, const float* __restrict__ Wout,
    const float* __restrict__ bout, float* __restrict__ out, int len)
{
    const int o = blockIdx.x;
    const float* w = Wout + (size_t)o * len;
    __shared__ float red[256];
    float s = 0.f;
    for (int i = threadIdx.x; i < len; i += 256) s += w[i] * (float)h[i];
    red[threadIdx.x] = s;
    __syncthreads();
    for (int off = 128; off > 0; off >>= 1) {
        if (threadIdx.x < off) red[threadIdx.x] += red[threadIdx.x + off];
        __syncthreads();
    }
    if (threadIdx.x == 0) out[o] = red[0] + bout[o];
}

// ---------------------------------------------------------------------------
// Host orchestration.
// d_in: 0=x, 1..5=no0..no4, base=6+8b: {W1,b1,g1,be1,W2,b2,g2,be2}_b, 46=Wout, 47=bout
// ---------------------------------------------------------------------------
extern "C" void kernel_launch(void* const* d_in, const int* in_sizes, int n_in,
                              void* d_out, int out_size, void* d_ws, size_t ws_size,
                              hipStream_t stream)
{
    (void)in_sizes; (void)n_in; (void)out_size; (void)ws_size;

    const float* x = (const float*)d_in[0];
    const int* no[5];
    for (int i = 0; i < 5; ++i) no[i] = (const int*)d_in[1 + i];
    auto P = [&](int i) { return (const float*)d_in[i]; };

    // workspace layout (all region sizes keep >=32B alignment)
    const size_t HBUF = (size_t)40962 * 64;           // max activation elems
    __bf16* hbfA = (__bf16*)d_ws;                     // bf16 ping
    __bf16* hbfB = hbfA + HBUF;                       // bf16 pong
    __bf16* wbf  = hbfB + HBUF;                       // bf16 weights (3,677,184)
    size_t wtot = 0;
    size_t woff[10];
    for (int b = 0; b < 5; ++b) {
        int cin = kCHS[b], cout = kCHS[b + 1];
        woff[2 * b]     = wtot; wtot += (size_t)cout * 7 * cin;
        woff[2 * b + 1] = wtot; wtot += (size_t)cout * 7 * cout;
    }
    float* yF    = (float*)(wbf + wtot);              // f32 conv output (max 40962*32)
    float* stats = yF + (size_t)40962 * 32;           // 2*512 floats

    // --- one-time (per launch) bf16 conversions: input + all weights ---
    {
        int n = 40962 * 64;
        cvt_bf16<<<(n + 255) / 256, 256, 0, stream>>>(x, hbfA, n);
        for (int b = 0; b < 5; ++b) {
            int cin = kCHS[b], cout = kCHS[b + 1], base = 6 + b * 8;
            int n1 = cout * 7 * cin;
            cvt_bf16<<<(n1 + 255) / 256, 256, 0, stream>>>(P(base), wbf + woff[2 * b], n1);
            int n2 = cout * 7 * cout;
            cvt_bf16<<<(n2 + 255) / 256, 256, 0, stream>>>(P(base + 4), wbf + woff[2 * b + 1], n2);
        }
    }

    const __bf16* h = hbfA;
    __bf16* alt = hbfB;
    __bf16* cur = hbfA;

    for (int b = 0; b < 5; ++b) {
        const int N    = kLEVELS[b];
        const int cin  = kCHS[b];
        const int cout = kCHS[b + 1];
        const int base = 6 + b * 8;

        if (b > 0) {
            int tot = N * cin;
            pool_mean7_bf16<<<(tot + 255) / 256, 256, 0, stream>>>(h, no[b - 1], alt, N, cin);
            h = alt; __bf16* t = alt; alt = cur; cur = t;
        }

        // conv1 + BN + lrelu
        {
            dim3 grid((N + 15) / 16, (cout + 127) / 128);
            conv_onering_wmma<<<grid, 256, 0, stream>>>(h, no[b], wbf + woff[2 * b],
                                                        P(base + 1), yF, N, cin, cout);
            bn_stats<<<cout, 256, 0, stream>>>(yF, N, cout, stats);
            int tot = N * cout;
            bn_lrelu_bf16<<<(tot + 255) / 256, 256, 0, stream>>>(yF, alt, N, cout, stats,
                                                                 P(base + 2), P(base + 3));
            h = alt; __bf16* t = alt; alt = cur; cur = t;
        }
        // conv2 + BN + lrelu
        {
            dim3 grid((N + 15) / 16, (cout + 127) / 128);
            conv_onering_wmma<<<grid, 256, 0, stream>>>(h, no[b], wbf + woff[2 * b + 1],
                                                        P(base + 5), yF, N, cout, cout);
            bn_stats<<<cout, 256, 0, stream>>>(yF, N, cout, stats);
            int tot = N * cout;
            bn_lrelu_bf16<<<(tot + 255) / 256, 256, 0, stream>>>(yF, alt, N, cout, stats,
                                                                 P(base + 6), P(base + 7));
            h = alt; __bf16* t = alt; alt = cur; cur = t;
        }
    }

    // final linear: [2, 162*512] @ h.flatten
    outc_dot<<<2, 256, 0, stream>>>(h, P(46), P(47), (float*)d_out, 162 * 512);
}
